// MIGPlannerV3_20220706030228
// MI455X (gfx1250) — matrix operand order repaired and checked by reading, not compiled, not run on hardware
//
#include <hip/hip_runtime.h>
#include <hip/hip_bf16.h>

// ---------------------------------------------------------------------------
// 2-layer edge-featured GAT, MI455X (gfx1250, wave32).
//   * attention logits folded to per-node scalar + 4-coef edge fold
//   * segment-max via uint-ordered-key atomicMax, segment-sum via f32 atomics
//   * layer-2 node projection [50000x128]x[128x128] via v_wmma_f32_16x16x32_f16
//     (A staged in LDS as f16, B pre-packed in per-lane WMMA operand order)
// ---------------------------------------------------------------------------

typedef __attribute__((ext_vector_type(16))) _Float16 v16h;
typedef __attribute__((ext_vector_type(8)))  float    v8f;

#define NN    50000
#define HID   128
#define NEDGE 1000000

// ---- float <-> monotone unsigned key (for atomic max over floats) ---------
__device__ __forceinline__ unsigned fkey(float f) {
    unsigned u = __float_as_uint(f);
    return (u & 0x80000000u) ? ~u : (u | 0x80000000u);
}
__device__ __forceinline__ float fkey_inv(unsigned k) {
    unsigned u = (k & 0x80000000u) ? (k & 0x7FFFFFFFu) : ~k;
    return __uint_as_float(u);
}
__device__ __forceinline__ float eluf(float v) {
    return v > 0.0f ? v : expm1f(v);
}

// ---- fold We,be against att: ec[0..3] = We^T att, ec[4] = be.att ----------
__global__ void edge_coef_k(const float* __restrict__ We,
                            const float* __restrict__ be,
                            const float* __restrict__ att,
                            float* __restrict__ ec) {
    int t = threadIdx.x;
    if (t < 4) {
        float s = 0.f;
        for (int c = 0; c < HID; ++c) s += We[t * HID + c] * att[c];
        ec[t] = s;
    } else if (t == 4) {
        float s = 0.f;
        for (int c = 0; c < HID; ++c) s += be[c] * att[c];
        ec[4] = s;
    }
}

// ---- layer-1 projection: h = x @ W1 + b1  (K = 2, pure streaming) ---------
__global__ __launch_bounds__(256) void project1_k(const float* __restrict__ x,
                                                  const float* __restrict__ W1,
                                                  const float* __restrict__ b1,
                                                  float* __restrict__ h) {
    int idx = blockIdx.x * 256 + threadIdx.x;   // over NN*HID
    int n = idx >> 7, c = idx & 127;
    if (n < NN)
        h[idx] = x[n * 2 + 0] * W1[c] + x[n * 2 + 1] * W1[HID + c] + b1[c];
}

// ---- per-node attention scalar: a[n] = h[n] . att  (wave per node) --------
__global__ __launch_bounds__(256) void node_dot_k(const float* __restrict__ h,
                                                  const float* __restrict__ att,
                                                  float* __restrict__ a) {
    int node = (blockIdx.x * 256 + threadIdx.x) >> 5;
    int lane = threadIdx.x & 31;
    if (node >= NN) return;
    const float* hp = h + (size_t)node * HID;
    float s = 0.f;
#pragma unroll
    for (int i = 0; i < 4; ++i) {
        int c = lane + i * 32;
        s = fmaf(hp[c], att[c], s);
    }
#pragma unroll
    for (int off = 16; off > 0; off >>= 1) s += __shfl_xor(s, off, 32);
    if (lane == 0) a[node] = s;
}

__global__ __launch_bounds__(256) void init_nodes_k(unsigned* __restrict__ amaxu,
                                                    float* __restrict__ denom) {
    int n = blockIdx.x * 256 + threadIdx.x;
    if (n < NN) { amaxu[n] = 0u; denom[n] = 0.f; }
}

// ---- per-edge alpha + segment max ----------------------------------------
__global__ __launch_bounds__(256) void edge_alpha_k(const long long* __restrict__ ei,
                                                    const float* __restrict__ ea,
                                                    const float* __restrict__ a,
                                                    const float* __restrict__ ec,
                                                    float* __restrict__ alpha,
                                                    unsigned* __restrict__ amaxu) {
    int e = blockIdx.x * 256 + threadIdx.x;
    if (e >= NEDGE) return;
    int s = (int)ei[e];
    int d = (int)ei[NEDGE + e];
    const float4 v = *reinterpret_cast<const float4*>(ea + (size_t)e * 4);
    float al = a[s] + a[d] + v.x * ec[0] + v.y * ec[1] + v.z * ec[2] + v.w * ec[3] + ec[4];
    if (v.y == 1.0f) al -= 1e9f;          // oneway mask
    alpha[e] = al;
    atomicMax(&amaxu[d], fkey(al));
}

// ---- per-edge exp + segment sum ------------------------------------------
__global__ __launch_bounds__(256) void edge_exp_k(const long long* __restrict__ ei,
                                                  const float* __restrict__ alpha,
                                                  const unsigned* __restrict__ amaxu,
                                                  float* __restrict__ exv,
                                                  float* __restrict__ denom) {
    int e = blockIdx.x * 256 + threadIdx.x;
    if (e >= NEDGE) return;
    int d = (int)ei[NEDGE + e];
    float ex = __expf(alpha[e] - fkey_inv(amaxu[d]));
    exv[e] = ex;
    atomicAdd(&denom[d], ex);
}

// ---- weighted scatter aggregation: agg[d] += h[s]*w  (wave per edge) ------
__global__ __launch_bounds__(256) void edge_agg_k(const long long* __restrict__ ei,
                                                  const float* __restrict__ exv,
                                                  const float* __restrict__ denom,
                                                  const float* __restrict__ feat,
                                                  float* __restrict__ agg) {
    long long wid = ((long long)blockIdx.x * 256 + threadIdx.x) >> 5;
    int lane = threadIdx.x & 31;
    if (wid >= NEDGE) return;
    int e = (int)wid;
    int s = (int)ei[e];
    int d = (int)ei[NEDGE + e];
    float w = exv[e] / (denom[d] + 1e-16f);
    const float* hp = feat + (size_t)s * HID;
    float* op = agg + (size_t)d * HID;
#pragma unroll
    for (int i = 0; i < 4; ++i) {
        int c = lane + i * 32;
        atomicAdd(op + c, hp[c] * w);
    }
}

// ---- in-place ELU ---------------------------------------------------------
__global__ __launch_bounds__(256) void elu_k(float* __restrict__ v) {
    int idx = blockIdx.x * 256 + threadIdx.x;
    if (idx < NN * HID) v[idx] = eluf(v[idx]);
}

// ---- pack W2 (f32 [128][128]) into per-lane WMMA B-operand order (f16) ----
// layout: pack[((nt*4 + kt)*32 + lane)*16 + i] = (f16) W2[k][n]
//   n = nt*16 + (lane&15);  k = kt*32 + 16*(lane>>4) + i     (i = 0..15)
__global__ __launch_bounds__(256) void pack_w2_k(const float* __restrict__ W2,
                                                 _Float16* __restrict__ pack) {
    int idx = blockIdx.x * 256 + threadIdx.x;   // 16384 total
    int i    = idx & 15;
    int lane = (idx >> 4) & 31;
    int kt   = (idx >> 9) & 3;
    int nt   = idx >> 11;
    int n = nt * 16 + (lane & 15);
    int k = kt * 32 + 16 * (lane >> 4) + i;
    pack[idx] = (_Float16)W2[k * HID + n];
}

// ---- layer-2 projection via WMMA: outh = act @ W2 + b2 --------------------
// block = 16 rows of M x all 128 N; 8 waves, one 16x16 N-tile each.
__global__ __launch_bounds__(256) void gemm_wmma_k(const float* __restrict__ act,
                                                   const _Float16* __restrict__ pack,
                                                   const float* __restrict__ b2,
                                                   float* __restrict__ outh) {
    __shared__ _Float16 ldsA[16 * 136];          // padded rows -> conflict-free
    const int m0  = blockIdx.x * 16;
    const int tid = threadIdx.x;

    // stage A tile (16x128) to LDS as f16, shared by all 8 waves
    for (int idx = tid; idx < 16 * 128; idx += 256) {
        int m = idx >> 7, c = idx & 127;
        ldsA[m * 136 + c] = (_Float16)act[(size_t)(m0 + m) * HID + c];
    }
    __syncthreads();

    const int wave = tid >> 5;        // n-tile index 0..7
    const int lane = tid & 31;
    const int hi   = lane >> 4;       // half-wave
    const int ml   = lane & 15;

    v8f acc = {};
#pragma unroll
    for (int kt = 0; kt < 4; ++kt) {
        // A operand: element i -> K = 16*hi + i (contiguous, matches packed B)
        v16h a;
#pragma unroll
        for (int i = 0; i < 16; ++i) {
            int kl = i + 16 * hi;   // 0..15 (hi=0), 16..31 (hi=1)
            a[i] = ldsA[ml * 136 + kt * 32 + kl];
        }
        // B operand: one contiguous 32B chunk per lane (pre-packed)
        v16h b = *reinterpret_cast<const v16h*>(pack + (((wave * 4 + kt) * 32 + lane) << 4));
        acc = __builtin_amdgcn_wmma_f32_16x16x32_f16(
            /*neg_a=*/false, a, /*neg_b=*/false, b,
            /*c_mod=*/(short)0, acc, /*reuse_a=*/false, /*reuse_b=*/false);
    }

    // C/D layout: VGPR v -> row m0 + v + 8*hi, col = wave*16 + ml
    const int n = wave * 16 + ml;
    const float bias = b2[n];
#pragma unroll
    for (int v = 0; v < 8; ++v) {
        int m = m0 + v + 8 * hi;
        outh[(size_t)m * HID + n] = acc[v] + bias;
    }
}

// ---- final head: out[n] = elu(agg[n]) . Wh + bh  (wave per node) ----------
__global__ __launch_bounds__(256) void final_k(const float* __restrict__ agg,
                                               const float* __restrict__ Wh,
                                               const float* __restrict__ bh,
                                               float* __restrict__ out) {
    int node = (blockIdx.x * 256 + threadIdx.x) >> 5;
    int lane = threadIdx.x & 31;
    if (node >= NN) return;
    const float* ap = agg + (size_t)node * HID;
    float s = 0.f;
#pragma unroll
    for (int i = 0; i < 4; ++i) {
        int c = lane + i * 32;
        s = fmaf(eluf(ap[c]), Wh[c], s);
    }
#pragma unroll
    for (int off = 16; off > 0; off >>= 1) s += __shfl_xor(s, off, 32);
    if (lane == 0) out[node] = s + bh[0];
}

// ---------------------------------------------------------------------------
extern "C" void kernel_launch(void* const* d_in, const int* in_sizes, int n_in,
                              void* d_out, int out_size, void* d_ws, size_t ws_size,
                              hipStream_t stream) {
    const float*     x    = (const float*)d_in[0];
    const long long* ei   = (const long long*)d_in[1];
    const float*     ea   = (const float*)d_in[2];
    const float*     W1   = (const float*)d_in[3];
    const float*     b1   = (const float*)d_in[4];
    const float*     We1  = (const float*)d_in[5];
    const float*     be1  = (const float*)d_in[6];
    const float*     att1 = (const float*)d_in[7];
    const float*     W2   = (const float*)d_in[8];
    const float*     b2   = (const float*)d_in[9];
    const float*     We2  = (const float*)d_in[10];
    const float*     be2  = (const float*)d_in[11];
    const float*     att2 = (const float*)d_in[12];
    const float*     Wh   = (const float*)d_in[13];
    const float*     bh   = (const float*)d_in[14];
    float*           out  = (float*)d_out;

    // workspace layout (all offsets keep >=32B alignment)
    float* ws    = (float*)d_ws;
    float* A     = ws;                             // 6,400,000 f  (feat / h2)
    float* Bb    = A + (size_t)NN * HID;           // 6,400,000 f  (agg / act1)
    float* alpha = Bb + (size_t)NN * HID;          // 1,000,000 f
    float* exv   = alpha + NEDGE;                  // 1,000,000 f
    float* anode = exv + NEDGE;                    //    50,000 f
    float* denom = anode + NN;                     //    50,000 f
    float* ec    = denom + NN;                     //        16 f reserved
    unsigned* amaxu = (unsigned*)(ec + 16);        //    50,000 u32
    _Float16* pack  = (_Float16*)((float*)(amaxu + NN));  // 16,384 f16 (32 KB)

    const size_t featBytes = (size_t)NN * HID * sizeof(float);
    const int eb = (NEDGE + 255) / 256;            // edge-parallel blocks
    const int nb = (NN + 255) / 256;
    const int wb = (NN * 32 + 255) / 256;          // wave-per-node blocks (6250)
    const int fb = (NN * HID + 255) / 256;         // elementwise feature blocks

    // ---------------- layer 1 ----------------
    hipMemsetAsync(Bb, 0, featBytes, stream);                       // agg1 = 0
    edge_coef_k<<<1, 32, 0, stream>>>(We1, be1, att1, ec);
    project1_k<<<fb, 256, 0, stream>>>(x, W1, b1, A);               // h1 -> A
    node_dot_k<<<wb, 256, 0, stream>>>(A, att1, anode);
    init_nodes_k<<<nb, 256, 0, stream>>>(amaxu, denom);
    edge_alpha_k<<<eb, 256, 0, stream>>>(ei, ea, anode, ec, alpha, amaxu);
    edge_exp_k<<<eb, 256, 0, stream>>>(ei, alpha, amaxu, exv, denom);
    edge_agg_k<<<NEDGE / 8, 256, 0, stream>>>(ei, exv, denom, A, Bb); // agg1 -> Bb
    elu_k<<<fb, 256, 0, stream>>>(Bb);                              // act1 = elu(agg1)

    // ---------------- layer 2 projection (WMMA) ----------------
    pack_w2_k<<<64, 256, 0, stream>>>(W2, pack);
    gemm_wmma_k<<<NN / 16, 256, 0, stream>>>(Bb, pack, b2, A);      // h2 -> A

    // ---------------- layer 2 attention/aggregation ----------------
    hipMemsetAsync(Bb, 0, featBytes, stream);                       // agg2 = 0
    edge_coef_k<<<1, 32, 0, stream>>>(We2, be2, att2, ec);
    node_dot_k<<<wb, 256, 0, stream>>>(A, att2, anode);
    init_nodes_k<<<nb, 256, 0, stream>>>(amaxu, denom);
    edge_alpha_k<<<eb, 256, 0, stream>>>(ei, ea, anode, ec, alpha, amaxu);
    edge_exp_k<<<eb, 256, 0, stream>>>(ei, alpha, amaxu, exv, denom);
    edge_agg_k<<<NEDGE / 8, 256, 0, stream>>>(ei, exv, denom, A, Bb);

    // ---------------- head ----------------
    final_k<<<wb, 256, 0, stream>>>(Bb, Wh, bh, out);
    (void)in_sizes; (void)n_in; (void)out_size; (void)ws_size;
}
